// Decoder_5248450035760
// MI455X (gfx1250) — compile-verified, matrix operand order checked by reading, not run on hardware
//
#include <hip/hip_runtime.h>
#include <hip/hip_bf16.h>

#define DEVI __device__ __forceinline__

typedef __attribute__((ext_vector_type(16))) __bf16   bf16x16;
typedef __attribute__((ext_vector_type(8)))  float    f32x8;
typedef __attribute__((ext_vector_type(4)))  unsigned uint4e;
typedef __attribute__((ext_vector_type(4)))  unsigned u32x4;
typedef __attribute__((ext_vector_type(8)))  int      i32x8;
typedef __attribute__((ext_vector_type(4)))  int      i32x4;

namespace {
constexpr int BB  = 128;      // batch
constexpr int TT  = 512;      // time steps
constexpr int HH  = 1024;     // hidden
constexpr int GG  = 4 * HH;   // gate dim (i,f,g,o)
constexpr int KT  = HH / 32;  // K-tiles of 32 per 1024 reduction
constexpr int NBL = 64;       // blocks per layer (each owns 16 hidden cols)
constexpr int NB  = 2 * NBL;  // total persistent workgroups (layer0 + layer1)
constexpr int NTH = 256;      // 8 wave32 per workgroup
// LDS layout (ushort elements); dynamic-LDS region assumed to start at byte 0
constexpr int LDSW0 = 0;          // L0: Whh0 tiles | L1: Wih1 tiles (128 KB)
constexpr int LDSW1 = 64 * 1024;  // L1: Whh1 tiles (128 KB)
constexpr unsigned SMEM_BYTES = 256 * 1024;
}

extern __shared__ unsigned short smem[];

// float -> bf16 round-to-nearest-even (bit trick; inputs are finite)
DEVI unsigned short f2bf(float f) {
  unsigned u = __builtin_bit_cast(unsigned, f);
  u += 0x7FFFu + ((u >> 16) & 1u);
  return (unsigned short)(u >> 16);
}
DEVI float sigm(float x) { return 1.0f / (1.0f + __expf(-x)); }
DEVI float tanh_(float x) {
  float ax = fabsf(x);
  float e  = __expf(-2.0f * ax);
  float t  = (1.0f - e) / (1.0f + e);
  return copysignf(t, x);
}
// butterfly sum across each 16-lane half of the wave32 (ds_swizzle xor mode)
DEVI float swz_add16(float v) {
  int x;
  x = __builtin_amdgcn_ds_swizzle(__builtin_bit_cast(int, v), 0x041F); v += __builtin_bit_cast(float, x);
  x = __builtin_amdgcn_ds_swizzle(__builtin_bit_cast(int, v), 0x081F); v += __builtin_bit_cast(float, x);
  x = __builtin_amdgcn_ds_swizzle(__builtin_bit_cast(int, v), 0x101F); v += __builtin_bit_cast(float, x);
  x = __builtin_amdgcn_ds_swizzle(__builtin_bit_cast(int, v), 0x201F); v += __builtin_bit_cast(float, x);
  return v;
}

union AFrag { uint4e q[2]; bf16x16 v; };

// acc[g] += A(h rows, K=1024) x B(LDS-resident W^T fragment tiles, gate g).
// A: bf16 row-major activations from global (L2-hot); B: ds_load_b128 pairs.
DEVI void gemm_gates_lds(f32x8 acc[4], const unsigned short* __restrict__ hsrc,
                         int wbase, int arow, int koff, int lane) {
  const unsigned short* arowp = hsrc + arow * HH;
  #pragma unroll 4
  for (int kt = 0; kt < KT; ++kt) {
    AFrag a;
    const uint4e* ap = reinterpret_cast<const uint4e*>(arowp + kt * 32 + koff);
    a.q[0] = ap[0];                 // K = koff .. koff+7
    a.q[1] = ap[2];                 // K = koff+16 .. koff+23
    #pragma unroll
    for (int g = 0; g < 4; ++g) {
      const uint4e* bp = reinterpret_cast<const uint4e*>(
          &smem[wbase + ((g * KT + kt) * 32 + lane) * 16]);
      AFrag b;
      b.q[0] = bp[0];
      b.q[1] = bp[1];
      acc[g] = __builtin_amdgcn_wmma_f32_16x16x32_bf16(
          false, a.v, false, b.v, (short)0, acc[g], false, false);
    }
  }
}

// Stage this block's 4 gate chunks (4 x 32KB, 2MB apart) of one packed weight
// matrix into LDS. Preferred path: one TDM tensor_load_to_lds descriptor with
// iterate_enable (4 iterations, lds+32KB / global+2MB per iteration).
#if __has_builtin(__builtin_amdgcn_tensor_load_to_lds)
#define HAVE_TDM 1
DEVI void tdm_stage(const unsigned short* gsrc, unsigned ldsByteOff) {
  unsigned long long ga = (unsigned long long)(uintptr_t)gsrc;
  u32x4 g0 = { 1u,                                   // count=1 (valid user D#)
               ldsByteOff,                           // lds_addr (bytes)
               (unsigned)(ga & 0xFFFFFFFFu),         // global_addr[31:0]
               (unsigned)((ga >> 32) & 0x01FFFFFFu)  // global_addr[56:32]
                   | (2u << 30) };                   // type=2 ("image")
  i32x8 g1 = { (3 << 16) | (1 << 19),  // data_size=8B, iterate_enable=1
               (int)(4096u << 16),     // tensor_dim0 = 4096 (8B units)
               (1 << 16),              // tensor_dim1 = 1
               (int)(4096u << 16),     // tile_dim0 = 4096 (32KB per iter)
               0,                      // tile_dim1 = tile_dim2 = 0 (1-D tile)
               4096, 0, 0 };           // tensor_dim0_stride = 4096
  i32x4 g2 = { 0,                      // tensor_dim2 (unused)
               4096,                   // lds_addr_increment (8B units) = 32KB
               262144,                 // global_addr_increment = 2MB / 8B
               (3 << 16) };            // iterate_count = 3 -> 4 iterations
  i32x4 g3 = { 0, 0, 0, 0 };
#if __clang_major__ >= 23
  i32x8 g4 = { 0, 0, 0, 0, 0, 0, 0, 0 };
  __builtin_amdgcn_tensor_load_to_lds(g0, g1, g2, g3, g4, 0);
#else
  __builtin_amdgcn_tensor_load_to_lds(g0, g1, g2, g3, 0);
#endif
}
#else
DEVI void fallback_stage(const unsigned short* gsrc, unsigned ldsElemOff, int tid) {
  #pragma unroll
  for (int g = 0; g < 4; ++g) {
    const uint4e* s = reinterpret_cast<const uint4e*>(gsrc + g * (64 * KT * 32 * 16));
    uint4e* d = reinterpret_cast<uint4e*>(&smem[ldsElemOff + g * 16384]);
    for (int i = tid; i < 2048; i += NTH) d[i] = s[i];   // 32 KB per gate chunk
  }
}
#endif

// Pack fp32 weight matrix [4096 x 1024] into bf16 WMMA B-fragment tiles:
// tile(ntile,ktile) = 32 lanes x 16 bf16, lane = n%16 + 16*(K-quadrant&1),
// elems 0-7 = low-K run, elems 8-15 = high-K run (matches A-layout symmetry).
__global__ void pack_weights(const float* __restrict__ Whh0,
                             const float* __restrict__ Wih1,
                             const float* __restrict__ Whh1,
                             unsigned short* __restrict__ Phh0,
                             unsigned short* __restrict__ Pih1,
                             unsigned short* __restrict__ Phh1) {
  int idx = blockIdx.x * blockDim.x + threadIdx.x;
  if (idx >= 3 * GG * HH) return;
  const int mat = idx / (GG * HH);
  const int rem = idx % (GG * HH);
  const int n = rem / HH, k = rem % HH;
  const float* src = (mat == 0) ? Whh0 : (mat == 1) ? Wih1 : Whh1;
  unsigned short* dst = (mat == 0) ? Phh0 : (mat == 1) ? Pih1 : Phh1;
  const int nt = n >> 4, nl = n & 15;
  const int kt = k >> 5, kk = k & 31;
  const int q = kk >> 3, r = kk & 7;
  const int lane = nl + (q & 1) * 16;
  const int elem = (q >> 1) * 8 + r;
  dst[((nt * KT + kt) * 32 + lane) * 16 + elem] = f2bf(src[n * HH + k]);
}

__global__ void init_state(const float* __restrict__ z,
                           unsigned short* h0buf, unsigned short* h1buf,
                           float* c0, float* c1, float* pred, unsigned* bar) {
  int i = blockIdx.x * blockDim.x + threadIdx.x;
  if (i < BB * HH) {
    float v = z[i];
    unsigned short b = f2bf(v);
    h0buf[i] = b;   // ping buffer 0
    h1buf[i] = b;
    c0[i] = v;
    c1[i] = v;
  }
  if (i < BB) pred[i] = 0.0f;
  if (i < 2)  bar[i] = 0u;
}

// Persistent 2-layer LSTM. Blocks [0,64) own layer 0, [64,128) own layer 1;
// layer-1's recurrent GEMM overlaps layer-0's cell (it depends only on t-1
// state), so the per-step critical path is 2 dependent K=1024 GEMMs, not 3.
// All weights are LDS-resident bf16 fragments staged once via the TDM.
__global__ void __launch_bounds__(NTH)
lstm_persistent(const float* __restrict__ seq,
                const float* __restrict__ Wih0,
                const float* __restrict__ bih0, const float* __restrict__ bhh0,
                const float* __restrict__ bih1, const float* __restrict__ bhh1,
                const float* __restrict__ Wout, const float* __restrict__ bout,
                const unsigned short* __restrict__ Phh0,
                const unsigned short* __restrict__ Pih1,
                const unsigned short* __restrict__ Phh1,
                unsigned short* h0buf, unsigned short* h1buf,
                float* c0, float* c1, float* pred, float* predpart,
                unsigned* bar, float* out) {
  const int tid    = threadIdx.x;
  const int lane   = tid & 31;
  const int wave   = tid >> 5;
  const int bid    = blockIdx.x;
  const bool isL0  = bid < NBL;
  const int bidL   = isL0 ? bid : bid - NBL;   // hidden-tile id within layer
  const int jl     = lane & 15;
  const int hiHalf = lane >> 4;
  const int m0     = wave * 16;                // row-tile base (batch rows)
  const int j      = bidL * 16 + jl;           // hidden column this lane owns
  const int arow   = m0 + jl;                  // A-fragment row (M = lane%16)
  const int koff   = hiHalf * 8;               // A/B K sub-offset per lane half
  const int rbase  = m0 + hiHalf * 8;          // C/D: M = vgpr + 8*(lane>=16)

  float bias0[4], bias1[4], w0g[4];
  #pragma unroll
  for (int g = 0; g < 4; ++g) {
    bias0[g] = bih0[g * HH + j] + bhh0[g * HH + j];
    bias1[g] = bih1[g * HH + j] + bhh1[g * HH + j];
    w0g[g]   = Wih0[g * HH + j];       // layer-0 input size F == 1
  }
  const float woutj = Wout[j];
  const float boutv = bout[0];

  // ---- stage this block's weights into LDS (once; step-invariant) --------
#ifdef HAVE_TDM
  if (wave == 0) {
    if (isL0) {
      tdm_stage(Phh0 + bidL * 16384, 0u);
    } else {
      tdm_stage(Pih1 + bidL * 16384, 0u);
      tdm_stage(Phh1 + bidL * 16384, (unsigned)(LDSW1 * 2));
    }
  }
  __builtin_amdgcn_s_wait_tensorcnt(0);
#else
  if (isL0) {
    fallback_stage(Phh0 + bidL * 16384, LDSW0, tid);
  } else {
    fallback_stage(Pih1 + bidL * 16384, LDSW0, tid);
    fallback_stage(Phh1 + bidL * 16384, LDSW1, tid);
  }
#endif
  __syncthreads();

  unsigned sensev = 0;
  auto gbar = [&]() {                  // device-wide sense-reversal barrier
    __syncthreads();
    if (tid == 0) {
      unsigned next = sensev + 1;
      __threadfence();
      unsigned prev = __hip_atomic_fetch_add(&bar[0], 1u, __ATOMIC_ACQ_REL,
                                             __HIP_MEMORY_SCOPE_AGENT);
      if (prev == (unsigned)(NB - 1)) {
        __hip_atomic_store(&bar[0], 0u, __ATOMIC_RELAXED, __HIP_MEMORY_SCOPE_AGENT);
        __hip_atomic_store(&bar[1], next, __ATOMIC_RELEASE, __HIP_MEMORY_SCOPE_AGENT);
      } else {
        while (__hip_atomic_load(&bar[1], __ATOMIC_ACQUIRE,
                                 __HIP_MEMORY_SCOPE_AGENT) != next)
          __builtin_amdgcn_s_sleep(1);
      }
    }
    __syncthreads();
    ++sensev;
  };

  float lossAcc = 0.0f;

  for (int t = 0; t < TT; ++t) {
    const int p = t & 1;
    const unsigned short* h0cur = h0buf + p * (BB * HH);
    unsigned short*       h0nxt = h0buf + (1 - p) * (BB * HH);
    const unsigned short* h1cur = h1buf + p * (BB * HH);
    unsigned short*       h1nxt = h1buf + (1 - p) * (BB * HH);

    f32x8 acc[4] = {};

    // -------- phase A (concurrent): L0 full cell | L1 recurrent GEMM ------
    if (isL0) {
      gemm_gates_lds(acc, h0cur, LDSW0, arow, koff, lane);
      #pragma unroll
      for (int e = 0; e < 8; ++e) {
        const int   b  = rbase + e;
        const float pv = pred[b];
        const float gi = acc[0][e] + bias0[0] + pv * w0g[0];
        const float gf = acc[1][e] + bias0[1] + pv * w0g[1];
        const float gg = acc[2][e] + bias0[2] + pv * w0g[2];
        const float go = acc[3][e] + bias0[3] + pv * w0g[3];
        const float cp = c0[b * HH + j];
        const float cn = sigm(gf) * cp + sigm(gi) * tanh_(gg);
        const float hn = sigm(go) * tanh_(cn);
        c0[b * HH + j]    = cn;
        h0nxt[b * HH + j] = f2bf(hn);
      }
    } else {
      gemm_gates_lds(acc, h1cur, LDSW1, arow, koff, lane);  // h1 @ Whh1^T
    }
    gbar();

    // -------- phase B: L1 input GEMM (new h0) + cell + proj partials ------
    if (!isL0) {
      gemm_gates_lds(acc, h0nxt, LDSW0, arow, koff, lane);  // h0_new @ Wih1^T
      #pragma unroll
      for (int e = 0; e < 8; ++e) {
        const int   b  = rbase + e;
        const float gi = acc[0][e] + bias1[0];
        const float gf = acc[1][e] + bias1[1];
        const float gg = acc[2][e] + bias1[2];
        const float go = acc[3][e] + bias1[3];
        const float cp = c1[b * HH + j];
        const float cn = sigm(gf) * cp + sigm(gi) * tanh_(gg);
        const float hn = sigm(go) * tanh_(cn);
        c1[b * HH + j]    = cn;
        h1nxt[b * HH + j] = f2bf(hn);
        const float ps = swz_add16(hn * woutj);  // sum this block's 16 j's
        if (jl == 0) predpart[bidL * BB + b] = ps;
      }
    }
    gbar();

    // -------- phase C: block 0 finishes pred, writes output, tracks loss --
    if (bid == 0 && tid < BB) {
      float s = boutv;
      #pragma unroll 8
      for (int blk = 0; blk < NBL; ++blk) s += predpart[blk * BB + tid];
      pred[tid]             = s;
      out[1 + tid * TT + t] = s;          // recovered[b, t, 0]
      const float d = seq[tid * TT + t] - s;
      lossAcc += d * d;
    }
    gbar();
  }

  // -------- final MSE reduction (block 0; scratch past its 128KB weights) --
  if (bid == 0) {
    float* red = reinterpret_cast<float*>(&smem[LDSW1]);
    red[tid] = lossAcc;
    __syncthreads();
    for (int s = NTH / 2; s > 0; s >>= 1) {
      if (tid < s) red[tid] += red[tid + s];
      __syncthreads();
    }
    if (tid == 0) out[0] = red[0] * (1.0f / (float)(BB * TT));
  }
}

extern "C" void kernel_launch(void* const* d_in, const int* in_sizes, int n_in,
                              void* d_out, int out_size, void* d_ws, size_t ws_size,
                              hipStream_t stream) {
  const float* seq  = (const float*)d_in[0];
  const float* z    = (const float*)d_in[1];
  const float* Wih0 = (const float*)d_in[2];
  const float* Whh0 = (const float*)d_in[3];
  const float* bih0 = (const float*)d_in[4];
  const float* bhh0 = (const float*)d_in[5];
  const float* Wih1 = (const float*)d_in[6];
  const float* Whh1 = (const float*)d_in[7];
  const float* bih1 = (const float*)d_in[8];
  const float* bhh1 = (const float*)d_in[9];
  const float* Wout = (const float*)d_in[10];
  const float* bout = (const float*)d_in[11];
  float* out = (float*)d_out;

  // Workspace carve-out (~26 MB): bf16-packed weights stay L2-resident.
  char* w = (char*)d_ws;
  auto carve = [&](size_t bytes) -> char* {
    char* p = w;
    w += (bytes + 255) & ~(size_t)255;
    return p;
  };
  unsigned short* Phh0  = (unsigned short*)carve((size_t)GG * HH * 2);
  unsigned short* Pih1  = (unsigned short*)carve((size_t)GG * HH * 2);
  unsigned short* Phh1  = (unsigned short*)carve((size_t)GG * HH * 2);
  unsigned short* h0buf = (unsigned short*)carve((size_t)2 * BB * HH * 2);
  unsigned short* h1buf = (unsigned short*)carve((size_t)2 * BB * HH * 2);
  float*    c0    = (float*)carve((size_t)BB * HH * 4);
  float*    c1    = (float*)carve((size_t)BB * HH * 4);
  float*    pred  = (float*)carve((size_t)BB * 4);
  float*    ppart = (float*)carve((size_t)NBL * BB * 4);
  unsigned* bar   = (unsigned*)carve(256);

  const int packTotal = 3 * GG * HH;
  pack_weights<<<(packTotal + 255) / 256, 256, 0, stream>>>(Whh0, Wih1, Whh1,
                                                            Phh0, Pih1, Phh1);
  init_state<<<(BB * HH + 255) / 256, 256, 0, stream>>>(z, h0buf, h1buf,
                                                        c0, c1, pred, bar);
  lstm_persistent<<<NB, NTH, SMEM_BYTES, stream>>>(seq, Wih0, bih0, bhh0,
                                                   bih1, bhh1, Wout, bout,
                                                   Phh0, Pih1, Phh1,
                                                   h0buf, h1buf, c0, c1,
                                                   pred, ppart, bar, out);
  (void)in_sizes; (void)n_in; (void)out_size; (void)ws_size;
}
